// RBFKANLayer_39883066311174
// MI455X (gfx1250) — compile-verified, matrix operand order checked by reading, not run on hardware
//
#include <hip/hip_runtime.h>
#include <hip/hip_bf16.h>
#include <stdint.h>

// RBF-KAN fused kernel for gfx1250 (MI455X).
//   out[B,O] = exp(-max(||x||^2 + ||c||^2 - 2 x@c^T, 0)) @ W
// B=65536, D=64, C=1024, O=256.
// Prep kernel: weights -> bf16 transposed [O][C], centers -> bf16, ||c||^2 f32 (in d_ws).
// Main kernel: both GEMMs via v_wmma_f32_16x16x32_bf16; double-buffered LDS chunks
// filled with global_load_async_to_lds_b128 (ASYNCcnt, inline asm) overlapping WMMAs.

typedef __attribute__((ext_vector_type(16))) __bf16 v16bf;
typedef __attribute__((ext_vector_type(8)))  __bf16 v8bf;
typedef __attribute__((ext_vector_type(4)))  __bf16 v4bf;
typedef __attribute__((ext_vector_type(8)))  float  v8f;

#define BM      128   // rows per block
#define CCHUNK  32    // centers per chunk (K of GEMM2, N of GEMM1)
#define DDIM    64
#define CDIM    1024
#define ODIM    256
#define XS_STRIDE 72  // 64 + 8 pad (bf16 elems) -> 144B rows, 16B aligned
#define CS_STRIDE 72
#define WT_STRIDE 40  // 32 + 8 pad -> 80B rows, 16B aligned
#define BS_STRIDE 40

static __device__ inline v16bf cat8(v8bf lo, v8bf hi) {
  return __builtin_shufflevector(lo, hi, 0,1,2,3,4,5,6,7,8,9,10,11,12,13,14,15);
}

// 16-byte async global->LDS copy (GLOBAL_LOAD_ASYNC_TO_LDS_B128, GV mode).
// VDST VGPR = wave-relative LDS byte offset = low 32 bits of the generic LDS pointer.
static __device__ inline void copy16(const __bf16* __restrict__ gsrc, __bf16* ldst) {
  unsigned lds_off = (unsigned)(uintptr_t)ldst;
  unsigned long long gaddr = (unsigned long long)(uintptr_t)gsrc;
  asm volatile("global_load_async_to_lds_b128 %0, %1, off"
               :: "v"(lds_off), "v"(gaddr)
               : "memory");
}

static __device__ inline void wait_async_publish() {
  asm volatile("s_wait_asynccnt 0" ::: "memory");
}

// ---------------- prep kernel: one block per center/weight row c ----------------
__global__ __launch_bounds__(256) void rbf_prep(
    const float* __restrict__ centers,   // [C][D]
    const float* __restrict__ weights,   // [C][O]
    __bf16* __restrict__ wt,             // [O][C] bf16 (transposed)
    __bf16* __restrict__ cb,             // [C][D] bf16
    float*  __restrict__ cn)             // [C]
{
  const int c = (int)blockIdx.x;         // 0..1023
  const int t = (int)threadIdx.x;        // 0..255
  // transpose weights row c into column c of wt
  wt[(size_t)t * CDIM + c] = (__bf16)weights[(size_t)c * ODIM + t];
  __shared__ float red[DDIM];
  if (t < DDIM) {
    float v = centers[(size_t)c * DDIM + t];
    cb[(size_t)c * DDIM + t] = (__bf16)v;
    red[t] = v * v;
  }
  __syncthreads();
  if (t == 0) {
    float s = 0.f;
    #pragma unroll
    for (int k = 0; k < DDIM; ++k) s += red[k];
    cn[c] = s;
  }
}

// ---------------- main fused kernel ----------------
__global__ __launch_bounds__(512) void rbf_kan_fused(
    const float*  __restrict__ x,        // [B][D] f32
    const __bf16* __restrict__ wt,       // [O][C] bf16
    const __bf16* __restrict__ cb,       // [C][D] bf16
    const float*  __restrict__ cn,       // [C] f32
    float* __restrict__ out)             // [B][O] f32
{
  extern __shared__ char smem[];
  __bf16* xs   = (__bf16*)(smem);                     // [128][72]   18432B
  __bf16* cs0  = (__bf16*)(smem + 18432);             // [32][72]     4608B
  __bf16* cs1  = (__bf16*)(smem + 23040);             // [32][72]     4608B
  __bf16* wT0  = (__bf16*)(smem + 27648);             // [256][40]   20480B
  __bf16* wT1  = (__bf16*)(smem + 48128);             // [256][40]   20480B
  __bf16* bs   = (__bf16*)(smem + 68608);             // [128][40]   10240B
  float*  xnorm = (float*)(smem + 78848);             // [128]         512B
  float*  cns   = (float*)(smem + 79360);             // [1024]       4096B
  __bf16* csb[2] = { cs0, cs1 };
  __bf16* wTb[2] = { wT0, wT1 };

  const int t    = (int)threadIdx.x;
  const int lane = t & 31;
  const int wave = t >> 5;      // 0..15
  const int mt   = wave >> 1;   // M-subtile 0..7 (16 rows each)
  const int nh   = wave & 1;    // N-half: columns [nh*128, nh*128+128)
  const int h    = lane >> 4;   // lane half (ISA A/B frag K split)
  const int n    = lane & 15;
  const int m0   = (int)blockIdx.x * BM;

  // ---- stage x tile f32 -> bf16 LDS (2048 float4 over 512 threads) ----
  const float4* xg4 = (const float4*)x;
  #pragma unroll
  for (int j = 0; j < 4; ++j) {
    int e  = j * 512 + t;
    int r  = e >> 4;
    int k4 = e & 15;
    float4 q = xg4[(size_t)(m0 + r) * 16 + k4];
    v4bf p = { (__bf16)q.x, (__bf16)q.y, (__bf16)q.z, (__bf16)q.w };
    *(v4bf*)&xs[r * XS_STRIDE + k4 * 4] = p;
  }
  // ---- x row norms in f32 ----
  if (t < BM) {
    const float4* row = (const float4*)(x + (size_t)(m0 + t) * DDIM);
    float s = 0.f;
    #pragma unroll
    for (int k = 0; k < 16; ++k) {
      float4 q = row[k];
      s += q.x*q.x + q.y*q.y + q.z*q.z + q.w*q.w;
    }
    xnorm[t] = s;
  }
  // ---- center norms -> LDS (2 per thread) ----
  cns[t]       = cn[t];
  cns[t + 512] = cn[t + 512];

  // ---- prologue: issue chunk 0 copy into buffer 0 ----
  {
    #pragma unroll
    for (int j = 0; j < 2; ++j) {          // weights chunk: 1024 x 16B
      int e = j * 512 + t;
      int o = e >> 2, seg = e & 3;
      copy16(wt + (size_t)o * CDIM + seg * 8, wTb[0] + o * WT_STRIDE + seg * 8);
    }
    if (t < 256) {                          // centers chunk: 256 x 16B
      int c = t >> 3, seg = t & 7;
      copy16(cb + (size_t)c * DDIM + seg * 8, csb[0] + c * CS_STRIDE + seg * 8);
    }
  }

  float xn[8];
  v8f acc[8] = {};

  for (int i = 0; i < CDIM / CCHUNK; ++i) {
    const int b  = i & 1;
    const int c0 = i * CCHUNK;

    wait_async_publish();     // this wave's async copies into buffer b done
    __syncthreads();          // all waves' copies visible; prev chunk consumers done

    if (i == 0) {
      #pragma unroll
      for (int r = 0; r < 8; ++r) xn[r] = xnorm[mt * 16 + r + 8 * h];
    }

    // issue next chunk copy into the other buffer (overlaps with WMMAs below)
    if (i + 1 < CDIM / CCHUNK) {
      const int c1 = c0 + CCHUNK;
      #pragma unroll
      for (int j = 0; j < 2; ++j) {
        int e = j * 512 + t;
        int o = e >> 2, seg = e & 3;
        copy16(wt + (size_t)o * CDIM + c1 + seg * 8,
               wTb[b ^ 1] + o * WT_STRIDE + seg * 8);
      }
      if (t < 256) {
        int c = t >> 3, seg = t & 7;
        copy16(cb + (size_t)(c1 + c) * DDIM + seg * 8,
               csb[b ^ 1] + c * CS_STRIDE + seg * 8);
      }
    }

    // ---- GEMM1: S(16x16) = x_sub(16xD) @ cen_half^T, K = 64 (2 WMMA) ----
    v8f s = {};
    #pragma unroll
    for (int kc = 0; kc < 2; ++kc) {
      const int k0 = kc * 32;
      const __bf16* ar = &xs[(mt * 16 + n) * XS_STRIDE + k0];
      v16bf a = cat8(*(const v8bf*)(ar + 8 * h), *(const v8bf*)(ar + 16 + 8 * h));
      const __bf16* br = &csb[b][(nh * 16 + n) * CS_STRIDE + k0];
      v16bf bm = cat8(*(const v8bf*)(br + 8 * h), *(const v8bf*)(br + 16 + 8 * h));
      s = __builtin_amdgcn_wmma_f32_16x16x32_bf16(false, a, false, bm, (short)0, s,
                                                  false, false);
    }
    // ---- distances, clamp, exp, pack basis strip to LDS ----
    const float cnv = cns[c0 + nh * 16 + n];
    #pragma unroll
    for (int r = 0; r < 8; ++r) {
      float sq = xn[r] + cnv - 2.0f * s[r];
      sq = fmaxf(sq, 0.0f);
      float bval = __expf(-sq);             // ALPHA = 1.0
      bs[(mt * 16 + r + 8 * h) * BS_STRIDE + (nh * 16 + n)] = (__bf16)bval;
    }
    __syncthreads();   // basis strip complete (both halves) before GEMM2

    // ---- GEMM2: acc += basis(16x32) @ W_chunk(32x256), 8 WMMA/wave ----
    const __bf16* ar = &bs[(mt * 16 + n) * BS_STRIDE];
    v16bf a = cat8(*(const v8bf*)(ar + 8 * h), *(const v8bf*)(ar + 16 + 8 * h));
    #pragma unroll
    for (int nt = 0; nt < 8; ++nt) {
      const __bf16* br = &wTb[b][(nh * 128 + nt * 16 + n) * WT_STRIDE];
      v16bf bm = cat8(*(const v8bf*)(br + 8 * h), *(const v8bf*)(br + 16 + 8 * h));
      acc[nt] = __builtin_amdgcn_wmma_f32_16x16x32_bf16(false, a, false, bm, (short)0,
                                                        acc[nt], false, false);
    }
  }

  // ---- store out tile (D-layout: VGPR r -> row r + 8h, lane n -> col) ----
  #pragma unroll
  for (int nt = 0; nt < 8; ++nt) {
    const int col = nh * 128 + nt * 16 + n;
    #pragma unroll
    for (int r = 0; r < 8; ++r) {
      const int row = m0 + mt * 16 + r + 8 * h;
      out[(size_t)row * ODIM + col] = acc[nt][r];
    }
  }
}

extern "C" void kernel_launch(void* const* d_in, const int* in_sizes, int n_in,
                              void* d_out, int out_size, void* d_ws, size_t ws_size,
                              hipStream_t stream) {
  (void)in_sizes; (void)n_in; (void)ws_size; (void)out_size;
  const float* x       = (const float*)d_in[0];
  const float* centers = (const float*)d_in[1];
  const float* weights = (const float*)d_in[2];
  float* out           = (float*)d_out;

  // workspace layout: wt bf16 [256][1024] | cb bf16 [1024][64] | cn f32 [1024]
  __bf16* wt = (__bf16*)d_ws;                                   // 524288 B
  __bf16* cb = (__bf16*)((char*)d_ws + 524288);                 // 131072 B
  float*  cn = (float*)((char*)d_ws + 524288 + 131072);         //   4096 B

  rbf_prep<<<dim3(CDIM), dim3(256), 0, stream>>>(centers, weights, wt, cb, cn);

  const int B = 65536;
  const size_t lds_bytes = 83456;   // xs|cs x2|wT x2|bs|xnorm|cns
  rbf_kan_fused<<<dim3(B / BM), dim3(512), lds_bytes, stream>>>(x, wt, cb, cn, out);
}
